// RelPartialLearnableMultiHeadAttn_6700148982298
// MI455X (gfx1250) — compile-verified, matrix operand order checked by reading, not run on hardware
//
#include <hip/hip_runtime.h>

// ---------------------------------------------------------------------------
// Types for CDNA5 WMMA (gfx1250, wave32)
// ---------------------------------------------------------------------------
typedef __bf16 bf16_t;
typedef __attribute__((ext_vector_type(16))) __bf16 v16bf;
typedef __attribute__((ext_vector_type(4)))  __bf16 v4bf;
typedef __attribute__((ext_vector_type(8)))  float  v8f;

// Problem constants
#define QL    2048
#define BSZ   4
#define NH    16
#define DH    64
#define DM    1024
#define HH    1024           // NH*DH
#define ROWS  8192           // QL*BSZ
#define TH3   3072           // 3*H
#define SCL   0.125f         // 1/sqrt(64)
#define LNEPS 1e-5f

// ---------------------------------------------------------------------------
// Elementwise: f32 -> bf16 convert (grid-stride)
// ---------------------------------------------------------------------------
__global__ __launch_bounds__(256) void k_cvt_bf16(const float* __restrict__ in,
                                                  bf16_t* __restrict__ out, long n) {
  for (long i = (long)blockIdx.x * blockDim.x + threadIdx.x; i < n;
       i += (long)gridDim.x * blockDim.x)
    out[i] = (bf16_t)in[i];
}

// Q with per-head biases added (Qac = q + r_w_bias, Qbd = q + r_r_bias), bf16
__global__ __launch_bounds__(256) void k_qbias(const bf16_t* __restrict__ wh,
                                               const float* __restrict__ rwb,
                                               const float* __restrict__ rrb,
                                               bf16_t* __restrict__ qac,
                                               bf16_t* __restrict__ qbd) {
  const long total = (long)ROWS * HH;
  for (long idx = (long)blockIdx.x * blockDim.x + threadIdx.x; idx < total;
       idx += (long)gridDim.x * blockDim.x) {
    long row = idx >> 10;               // (i*BSZ + b)
    int  col = (int)(idx & 1023);       // n*64 + d
    float q = (float)wh[row * TH3 + col];
    qac[idx] = (bf16_t)(q + rwb[col]);
    qbd[idx] = (bf16_t)(q + rrb[col]);
  }
}

// V transpose: vt[((b*16+n)*64 + d)*2048 + j] = V[j,b,n,d]
__global__ __launch_bounds__(256) void k_vt(const bf16_t* __restrict__ wh,
                                            bf16_t* __restrict__ vt) {
  const long total = (long)BSZ * NH * DH * QL;   // 8,388,608
  for (long idx = (long)blockIdx.x * blockDim.x + threadIdx.x; idx < total;
       idx += (long)gridDim.x * blockDim.x) {
    int  j = (int)(idx & 2047);
    long rest = idx >> 11;
    int  d = (int)(rest & 63);  rest >>= 6;
    int  n = (int)(rest & 15);
    int  b = (int)(rest >> 4);
    vt[idx] = wh[((long)j * BSZ + b) * TH3 + 2 * HH + n * DH + d];
  }
}

// ---------------------------------------------------------------------------
// Generic bf16 TN GEMM:  C[M,N] = A[M,K] * B[N,K]^T   (f32 accumulate)
// Block = 128x128x32, 256 threads = 8 wave32 waves in a 2x4 grid,
// wave tile 64x32 = 4x2 WMMA(16x16x32) tiles. Double-buffered LDS.
// GUARD=false (all dims multiples of the tile): global->LDS staging uses
// CDNA5 GLOBAL_LOAD_ASYNC_TO_LDS_B128 tracked by ASYNCcnt, overlapped with
// the WMMA work on the other buffer. GUARD=true keeps the synchronous
// zero-filling path (needed for the N=64 P*V GEMM).
// ---------------------------------------------------------------------------
#define BM 128
#define BN 128
#define BK 32
#define BKP 40   // padded LDS row (bf16 elements) to spread banks

// 16x32 bf16 fragment from LDS (row-major, ld elements). Per CDNA5 ISA:
// lane L holds row m=L%16; half=L/16 selects K subgroup; VGPR v covers
// K = ((v&4)?16:0) + half*8 + (v&3)*2, two packed bf16 per VGPR.
__device__ inline v16bf lds_frag(const bf16_t* base, int ld, int row0) {
  int lane = threadIdx.x & 31;
  int m = lane & 15, half = lane >> 4;
  const bf16_t* r = base + (row0 + m) * ld;
  v16bf f;
#pragma unroll
  for (int v = 0; v < 8; ++v) {
    int k = ((v & 4) << 2) + half * 8 + ((v & 3) << 1);
    f[2 * v]     = r[k];
    f[2 * v + 1] = r[k + 1];
  }
  return f;
}

// Async copy of 16B global -> LDS (CDNA5 TDM-adjacent path, ASYNCcnt).
__device__ inline void async_cp16(bf16_t* lds, const bf16_t* g) {
  unsigned l = (unsigned)(unsigned long long)lds;   // low 32 bits = DS offset
  asm volatile("global_load_async_to_lds_b128 %0, %1, off"
               :: "v"(l), "v"(g) : "memory");
}
__device__ inline void async_wait0() {
  asm volatile("s_wait_asynccnt 0x0" ::: "memory");
}

template <bool GUARD>
__global__ __launch_bounds__(256)
void k_gemm_nt(const bf16_t* __restrict__ A, long lda,
               const bf16_t* __restrict__ B, long ldb,
               void* __restrict__ Cv, long ldc,
               int M, int N, int K, int c_is_f32) {
  __shared__ bf16_t As[2][BM * BKP];
  __shared__ bf16_t Bs[2][BN * BKP];

  const int tid  = threadIdx.x;
  const int lane = tid & 31;
  const int wid  = tid >> 5;
  const int wm   = wid >> 2;      // 0..1
  const int wn   = wid & 3;       // 0..3
  const int m0   = blockIdx.y * BM;
  const int n0   = blockIdx.x * BN;

  v8f acc[4][2] = {};

  // --- staging: async (unguarded) ---
  auto loadA_async = [&](int buf, int k0) {
#pragma unroll
    for (int t = 0; t < 2; ++t) {
      int ch  = tid + t * 256;          // 512 chunks of 8 bf16 (16B)
      int row = ch >> 2;
      int kc  = (ch & 3) << 3;
      async_cp16(&As[buf][row * BKP + kc], A + (long)(m0 + row) * lda + k0 + kc);
    }
  };
  auto loadB_async = [&](int buf, int k0) {
#pragma unroll
    for (int t = 0; t < 2; ++t) {
      int ch  = tid + t * 256;
      int row = ch >> 2;
      int kc  = (ch & 3) << 3;
      async_cp16(&Bs[buf][row * BKP + kc], B + (long)(n0 + row) * ldb + k0 + kc);
    }
  };
  // --- staging: synchronous with zero-fill (guarded) ---
  auto loadA_g = [&](int buf, int k0) {
#pragma unroll
    for (int t = 0; t < 4; ++t) {
      int ch  = tid + t * 256;          // 1024 chunks of 4 bf16
      int row = ch >> 3;
      int kc  = (ch & 7) << 2;
      v4bf val = {};
      int gr = m0 + row;
      if (gr < M) val = *(const v4bf*)(A + (long)gr * lda + k0 + kc);
      *(v4bf*)(&As[buf][row * BKP + kc]) = val;
    }
  };
  auto loadB_g = [&](int buf, int k0) {
#pragma unroll
    for (int t = 0; t < 4; ++t) {
      int ch  = tid + t * 256;
      int row = ch >> 3;
      int kc  = (ch & 7) << 2;
      v4bf val = {};
      int gr = n0 + row;
      if (gr < N) val = *(const v4bf*)(B + (long)gr * ldb + k0 + kc);
      *(v4bf*)(&Bs[buf][row * BKP + kc]) = val;
    }
  };

  const int nk = K / BK;
  if (GUARD) { loadA_g(0, 0); loadB_g(0, 0); }
  else       { loadA_async(0, 0); loadB_async(0, 0); async_wait0(); }
  __syncthreads();

  for (int kb = 0; kb < nk; ++kb) {
    int cur = kb & 1, nxt = cur ^ 1;
    const bool more = (kb + 1 < nk);
    if (more) {
      if (GUARD) { loadA_g(nxt, (kb + 1) * BK); loadB_g(nxt, (kb + 1) * BK); }
      else       { loadA_async(nxt, (kb + 1) * BK); loadB_async(nxt, (kb + 1) * BK); }
    }

    v16bf af[4], bfr[2];
#pragma unroll
    for (int mi = 0; mi < 4; ++mi)
      af[mi] = lds_frag(&As[cur][0], BKP, wm * 64 + mi * 16);
#pragma unroll
    for (int ni = 0; ni < 2; ++ni)
      bfr[ni] = lds_frag(&Bs[cur][0], BKP, wn * 32 + ni * 16);

#pragma unroll
    for (int mi = 0; mi < 4; ++mi)
#pragma unroll
      for (int ni = 0; ni < 2; ++ni)
        acc[mi][ni] = __builtin_amdgcn_wmma_f32_16x16x32_bf16(
            false, af[mi], false, bfr[ni], (short)0, acc[mi][ni], false, false);

    if (!GUARD && more) async_wait0();   // drain nxt-buffer copies before barrier
    __syncthreads();
  }

  // C/D layout: VGPR v, half=lane>>4 -> row = v + 8*half, col = lane&15
  const int half = lane >> 4, col = lane & 15;
#pragma unroll
  for (int mi = 0; mi < 4; ++mi)
#pragma unroll
    for (int ni = 0; ni < 2; ++ni) {
      int gn = n0 + wn * 32 + ni * 16 + col;
      if (GUARD && gn >= N) continue;
      int gmb = m0 + wm * 64 + mi * 16 + half * 8;
#pragma unroll
      for (int v = 0; v < 8; ++v) {
        int gm = gmb + v;
        if (GUARD && gm >= M) continue;
        if (c_is_f32) ((float*)Cv)[(long)gm * ldc + gn]  = acc[mi][ni][v];
        else          ((bf16_t*)Cv)[(long)gm * ldc + gn] = (bf16_t)acc[mi][ni][v];
      }
    }
}

// ---------------------------------------------------------------------------
// rel_shift + scale + softmax over k (one block per query row).
// shifted(i,j): u=j-i;  u<=0 -> BD[i, u+L-1];  u==1 -> 0;  u>=2 -> BD[i+1, u-2]
// ---------------------------------------------------------------------------
__global__ __launch_bounds__(256)
void k_softmax_shift(const float* __restrict__ AC, const float* __restrict__ BD,
                     bf16_t* __restrict__ P) {
  const int L = QL;
  const int i = blockIdx.x;
  const int t = threadIdx.x;
  __shared__ float red[256];

  float sv[8];
  float lmax = -3.4e38f;
#pragma unroll
  for (int c = 0; c < 8; ++c) {
    int j = t + c * 256;
    int u = j - i;
    float bd;
    if (u <= 0)      bd = BD[(long)i * L + (u + L - 1)];
    else if (u == 1) bd = 0.f;
    else             bd = BD[(long)(i + 1) * L + (u - 2)];
    float s = (AC[(long)i * L + j] + bd) * SCL;
    sv[c] = s;
    lmax = fmaxf(lmax, s);
  }
  red[t] = lmax; __syncthreads();
  for (int o = 128; o > 0; o >>= 1) {
    if (t < o) red[t] = fmaxf(red[t], red[t + o]);
    __syncthreads();
  }
  float mx = red[0]; __syncthreads();

  float lsum = 0.f;
#pragma unroll
  for (int c = 0; c < 8; ++c) { sv[c] = __expf(sv[c] - mx); lsum += sv[c]; }
  red[t] = lsum; __syncthreads();
  for (int o = 128; o > 0; o >>= 1) {
    if (t < o) red[t] += red[t + o];
    __syncthreads();
  }
  float inv = 1.0f / red[0];
#pragma unroll
  for (int c = 0; c < 8; ++c) {
    int j = t + c * 256;
    P[(long)i * L + j] = (bf16_t)(sv[c] * inv);
  }
}

// ---------------------------------------------------------------------------
// residual + LayerNorm, one block per (i,b) row of 1024
// ---------------------------------------------------------------------------
__global__ __launch_bounds__(256)
void k_residual_ln(const float* __restrict__ w, const float* __restrict__ ao,
                   const float* __restrict__ g, const float* __restrict__ b,
                   float* __restrict__ out) {
  const long row = blockIdx.x;
  const int  t   = threadIdx.x;
  __shared__ float red[256];

  float x[4]; float s = 0.f;
#pragma unroll
  for (int c = 0; c < 4; ++c) {
    int col = t + c * 256;
    x[c] = w[row * DM + col] + ao[row * DM + col];
    s += x[c];
  }
  red[t] = s; __syncthreads();
  for (int o = 128; o > 0; o >>= 1) { if (t < o) red[t] += red[t + o]; __syncthreads(); }
  float mu = red[0] * (1.0f / DM); __syncthreads();

  float v = 0.f;
#pragma unroll
  for (int c = 0; c < 4; ++c) { float d = x[c] - mu; v += d * d; }
  red[t] = v; __syncthreads();
  for (int o = 128; o > 0; o >>= 1) { if (t < o) red[t] += red[t + o]; __syncthreads(); }
  float rstd = rsqrtf(red[0] * (1.0f / DM) + LNEPS);

#pragma unroll
  for (int c = 0; c < 4; ++c) {
    int col = t + c * 256;
    out[row * DM + col] = (x[c] - mu) * rstd * g[col] + b[col];
  }
}

// ---------------------------------------------------------------------------
// Host orchestration
// ---------------------------------------------------------------------------
extern "C" void kernel_launch(void* const* d_in, const int* in_sizes, int n_in,
                              void* d_out, int out_size, void* d_ws, size_t ws_size,
                              hipStream_t stream) {
  (void)in_sizes; (void)n_in; (void)out_size; (void)ws_size;
  const float* w     = (const float*)d_in[0];
  const float* r     = (const float*)d_in[1];
  const float* qkvw  = (const float*)d_in[2];
  const float* rnetw = (const float*)d_in[3];
  const float* ow    = (const float*)d_in[4];
  const float* rwb   = (const float*)d_in[5];
  const float* rrb   = (const float*)d_in[6];
  const float* lng   = (const float*)d_in[7];
  const float* lnb   = (const float*)d_in[8];

  char* p = (char*)d_ws;
  auto take = [&](size_t bytes) -> char* {
    char* q = p; p += (bytes + 255) & ~(size_t)255; return q;
  };
  bf16_t* w_bf    = (bf16_t*)take((size_t)ROWS * DM * 2);        // 16.8 MB
  bf16_t* qkvw_bf = (bf16_t*)take((size_t)TH3 * DM * 2);         //  6.3 MB
  bf16_t* r_bf    = (bf16_t*)take((size_t)QL * DM * 2);          //  4.2 MB
  bf16_t* rnw_bf  = (bf16_t*)take((size_t)HH * DM * 2);          //  2.1 MB
  bf16_t* ow_bf   = (bf16_t*)take((size_t)DM * HH * 2);          //  2.1 MB
  bf16_t* wh      = (bf16_t*)take((size_t)ROWS * TH3 * 2);       // 50.3 MB  (QKV packed)
  bf16_t* rk      = (bf16_t*)take((size_t)QL * HH * 2);          //  4.2 MB
  bf16_t* qac     = (bf16_t*)take((size_t)ROWS * HH * 2);        // 16.8 MB
  bf16_t* qbd     = (bf16_t*)take((size_t)ROWS * HH * 2);        // 16.8 MB
  bf16_t* vt      = (bf16_t*)take((size_t)BSZ * NH * DH * QL * 2); // 16.8 MB
  float*  ACb     = (float*) take((size_t)QL * QL * 4);          // 16.8 MB (per-slice reuse)
  float*  BDb     = (float*) take((size_t)QL * QL * 4);          // 16.8 MB
  bf16_t* Pb      = (bf16_t*)take((size_t)QL * QL * 2);          //  8.4 MB
  bf16_t* av      = (bf16_t*)take((size_t)ROWS * HH * 2);        // 16.8 MB
  float*  aout    = (float*) take((size_t)ROWS * DM * 4);        // 33.6 MB

  auto gemm = [&](const bf16_t* A, long lda, const bf16_t* B, long ldb,
                  void* C, long ldc, int M, int N, int K, int cf32) {
    dim3 grid((N + BN - 1) / BN, (M + BM - 1) / BM);
    const bool guard = (M % BM) || (N % BN);
    if (guard)
      k_gemm_nt<true><<<grid, 256, 0, stream>>>(A, lda, B, ldb, C, ldc, M, N, K, cf32);
    else
      k_gemm_nt<false><<<grid, 256, 0, stream>>>(A, lda, B, ldb, C, ldc, M, N, K, cf32);
  };

  // 1) convert fp32 inputs to bf16
  k_cvt_bf16<<<1024, 256, 0, stream>>>(w,     w_bf,    (long)ROWS * DM);
  k_cvt_bf16<<<1024, 256, 0, stream>>>(qkvw,  qkvw_bf, (long)TH3 * DM);
  k_cvt_bf16<<<1024, 256, 0, stream>>>(r,     r_bf,    (long)QL * DM);
  k_cvt_bf16<<<1024, 256, 0, stream>>>(rnetw, rnw_bf,  (long)HH * DM);
  k_cvt_bf16<<<1024, 256, 0, stream>>>(ow,    ow_bf,   (long)DM * HH);

  // 2) QKV projection: wh[ROWS,3H] = w_bf[ROWS,DM] * qkvw_bf[3H,DM]^T
  gemm(w_bf, DM, qkvw_bf, DM, wh, TH3, ROWS, TH3, DM, 0);
  // 3) rel-pos keys: rk[QL,H] = r_bf * rnw_bf^T
  gemm(r_bf, DM, rnw_bf, DM, rk, HH, QL, HH, DM, 0);

  // 4) Q + biases; V transpose
  k_qbias<<<1024, 256, 0, stream>>>(wh, rwb, rrb, qac, qbd);
  k_vt<<<1024, 256, 0, stream>>>(wh, vt);

  // 5) per-(b, head) attention: AC, BD, shift+softmax, P*V
  for (int b = 0; b < BSZ; ++b)
    for (int n = 0; n < NH; ++n) {
      long qoff = (long)b * HH + (long)n * DH;                 // into qac/qbd/av, ld 4096
      const bf16_t* Kp = wh + (long)b * TH3 + HH + (long)n * DH;    // ldb = BSZ*3H
      const bf16_t* Rp = rk + (long)n * DH;                          // ldb = H
      const bf16_t* Vt = vt + ((long)(b * NH + n) * DH) * QL;        // ldb = QL

      gemm(qac + qoff, (long)BSZ * HH, Kp, (long)BSZ * TH3, ACb, QL, QL, QL, DH, 1);
      gemm(qbd + qoff, (long)BSZ * HH, Rp, HH,              BDb, QL, QL, QL, DH, 1);
      k_softmax_shift<<<QL, 256, 0, stream>>>(ACb, BDb, Pb);
      gemm(Pb, QL, Vt, QL, av + qoff, (long)BSZ * HH, QL, DH, QL, 0);
    }

  // 6) output projection: aout[ROWS,DM] = av[ROWS,H] * ow_bf[DM,H]^T  (f32 out)
  gemm(av, HH, ow_bf, HH, aout, DM, ROWS, DM, HH, 1);

  // 7) residual + LayerNorm -> d_out (f32, [qlen, bsz, d_model])
  k_residual_ln<<<ROWS, 256, 0, stream>>>(w, aout, lng, lnb, (float*)d_out);
}
